// CausalSelfAttention_403726926296
// MI455X (gfx1250) — matrix operand order repaired and checked by reading, not compiled, not run on hardware
//
#include <hip/hip_runtime.h>
#include <hip/hip_bf16.h>
#include <stdint.h>

// Problem constants (fixed by the reference).
#define B_   2
#define T_   2048
#define D_   2048
#define NH_  16
#define NKV_ 4
#define HD_  128
#define MROWS_ (B_ * T_)           // 4096
#define KVD_   (NKV_ * HD_)        // 512
#define EPS_   1.1920928955078125e-7f

typedef __bf16 bf16;
typedef __attribute__((ext_vector_type(16))) __bf16 v16bf;
typedef __attribute__((ext_vector_type(8)))  __bf16 v8bf;
typedef __attribute__((ext_vector_type(4)))  __bf16 v4bf;
typedef __attribute__((ext_vector_type(8)))  float  v8f;

static __device__ __forceinline__ bf16 f2bf(float f) {
    unsigned u = __float_as_uint(f);
    u += 0x7fffu + ((u >> 16) & 1u);           // round-to-nearest-even
    unsigned short h = (unsigned short)(u >> 16);
    bf16 r;
    __builtin_memcpy(&r, &h, sizeof(r));
    return r;
}

static __device__ __forceinline__ v16bf combine16(v8bf a, v8bf b) {
    v16bf r;
#pragma unroll
    for (int i = 0; i < 8; ++i) { r[i] = a[i]; r[i + 8] = b[i]; }
    return r;
}

// ---------------------------------------------------------------------------
// fp32 -> bf16 conversion, 4 elements per thread
// ---------------------------------------------------------------------------
__global__ void cvt_f32_bf16(const float4* __restrict__ in, v4bf* __restrict__ out, int n4) {
    int i = blockIdx.x * blockDim.x + threadIdx.x;
    if (i < n4) {
        float4 v = in[i];
        v4bf o;
        o[0] = f2bf(v.x); o[1] = f2bf(v.y); o[2] = f2bf(v.z); o[3] = f2bf(v.w);
        out[i] = o;
    }
}

// ---------------------------------------------------------------------------
// C[M,N] = A[M,K] * B[N,K]^T   (bf16 in, fp32 out).
// Register-blocked: each wave computes a 64x32 patch (4 M-tiles x 2 N-tiles,
// 8 WMMA accumulators). A 256-thread block (8 waves, 2 along M x 4 along N)
// covers a 128x128 macro-tile. Per 32-K step: 4 A-frags + 2 B-frags -> 8 WMMA.
//
// A-fragment (16x32 bf16): lane l holds row m=l%16; halves i<8 -> k=i+8*(l>=16),
// halves i>=8 -> k = 16+(i-8)+8*(l>=16)   => two contiguous 16B loads.
// B-fragment (32x16): lane l holds col n=l%16; same k order as A-fragment
//   => two contiguous 16B loads.
// Grid: (N/128, M/128).
// ---------------------------------------------------------------------------
__global__ __launch_bounds__(256) void gemm_bf16_nt(const bf16* __restrict__ A,
                                                    const bf16* __restrict__ Bw,
                                                    float* __restrict__ C,
                                                    int M, int N, int K) {
    const int lane = threadIdx.x & 31;
    const int wave = threadIdx.x >> 5;
    const int hi   = lane >> 4;
    const int l16  = lane & 15;

    const int m0 = blockIdx.y * 128 + (wave >> 2) * 64;   // wave: 4 M-tiles
    const int n0 = blockIdx.x * 128 + (wave & 3) * 32;    // wave: 2 N-tiles

    const bf16* arow[4];
    const bf16* brow[2];
#pragma unroll
    for (int i = 0; i < 4; ++i) arow[i] = A  + (size_t)(m0 + i * 16 + l16) * K;
#pragma unroll
    for (int j = 0; j < 2; ++j) brow[j] = Bw + (size_t)(n0 + j * 16 + l16) * K;

    v8f acc[4][2];
#pragma unroll
    for (int i = 0; i < 4; ++i)
#pragma unroll
        for (int j = 0; j < 2; ++j)
#pragma unroll
            for (int v = 0; v < 8; ++v) acc[i][j][v] = 0.0f;

    for (int k0 = 0; k0 < K; k0 += 32) {
        v16bf av[4], bv[2];
#pragma unroll
        for (int i = 0; i < 4; ++i) {
            v8bf a0 = *(const v8bf*)(arow[i] + k0 + hi * 8);
            v8bf a1 = *(const v8bf*)(arow[i] + k0 + 16 + hi * 8);
            av[i] = combine16(a0, a1);
        }
#pragma unroll
        for (int j = 0; j < 2; ++j) {
            v8bf b0 = *(const v8bf*)(brow[j] + k0 + hi * 8);
            v8bf b1 = *(const v8bf*)(brow[j] + k0 + 16 + hi * 8);
            bv[j] = combine16(b0, b1);
        }
#pragma unroll
        for (int i = 0; i < 4; ++i)
#pragma unroll
            for (int j = 0; j < 2; ++j)
                acc[i][j] = __builtin_amdgcn_wmma_f32_16x16x32_bf16(
                    false, av[i], false, bv[j], (short)0, acc[i][j], false, false);
    }

    // D layout: lane l holds col n=l%16, VGPR v holds row m = v + 8*(l>=16)
#pragma unroll
    for (int i = 0; i < 4; ++i)
#pragma unroll
        for (int j = 0; j < 2; ++j) {
            float* crow = C + (size_t)(m0 + i * 16 + hi * 8) * N + n0 + j * 16 + l16;
#pragma unroll
            for (int v = 0; v < 8; ++v) crow[(size_t)v * N] = acc[i][j][v];
        }
}

// ---------------------------------------------------------------------------
// RoPE + RMSNorm (mode 0) writing head-major bf16 [b][h][t][d], or
// V pass (mode 1): plain convert + transpose to [b][h][d][t] (d-major)
// One wave handles one 128-element head vector; lane owns 4 consecutive elems.
// src vector lives at src + (b*T+t)*rowstride + h*128.
// ---------------------------------------------------------------------------
__global__ __launch_bounds__(256) void rope_norm_kernel(const float* __restrict__ src,
                                                        int rowstride,
                                                        const float* __restrict__ normw,
                                                        bf16* __restrict__ dst,
                                                        int nheads, int mode, int nvec) {
    const int vec  = blockIdx.x * (blockDim.x >> 5) + (threadIdx.x >> 5);
    const int lane = threadIdx.x & 31;
    if (vec >= nvec) return;

    const int b   = vec / (T_ * nheads);
    const int rem = vec % (T_ * nheads);
    const int t   = rem / nheads;
    const int h   = rem % nheads;

    const float* p = src + (size_t)(b * T_ + t) * rowstride + h * HD_ + lane * 4;
    float4 xv = *(const float4*)p;
    float x[4] = {xv.x, xv.y, xv.z, xv.w};

    if (mode == 0) {
        // rope: pair i uses elements (2i, 2i+1); inv_freq = theta^(-2i/hd)
#pragma unroll
        for (int pr = 0; pr < 2; ++pr) {
            int i = lane * 2 + pr;                        // pair index 0..63
            float fr  = __powf(10000.0f, -((float)(2 * i)) / 128.0f);
            float ang = (float)t * fr;
            float s, c;
            __sincosf(ang, &s, &c);
            float xr = x[2 * pr], xi = x[2 * pr + 1];
            x[2 * pr]     = xr * c - xi * s;
            x[2 * pr + 1] = xr * s + xi * c;
        }
        // rmsnorm over 128 elems
        float ss = x[0] * x[0] + x[1] * x[1] + x[2] * x[2] + x[3] * x[3];
#pragma unroll
        for (int off = 16; off >= 1; off >>= 1) ss += __shfl_xor(ss, off, 32);
        float r = rsqrtf(ss * (1.0f / 128.0f) + EPS_);

        v4bf o;
#pragma unroll
        for (int e = 0; e < 4; ++e) o[e] = f2bf(x[e] * r * normw[lane * 4 + e]);
        bf16* d = dst + ((size_t)(b * nheads + h) * T_ + t) * HD_ + lane * 4;
        *(v4bf*)d = o;
    } else {
        // V: transpose store [b][h][d][t]
        size_t base = (size_t)(b * nheads + h) * HD_ * (size_t)T_;
#pragma unroll
        for (int e = 0; e < 4; ++e)
            dst[base + (size_t)(lane * 4 + e) * T_ + t] = f2bf(x[e]);
    }
}

// ---------------------------------------------------------------------------
// Flash attention: one wave owns a 16-row q tile, streams keys in chunks of 32.
// Qn [b][h][t][128] bf16, Kn [b][kvh][t][128] bf16, Vt [b][kvh][d][t] bf16.
// Output bf16 [b][t][h*128+d].
// ---------------------------------------------------------------------------
__global__ __launch_bounds__(128) void attn_kernel(const bf16* __restrict__ Qn,
                                                   const bf16* __restrict__ Kn,
                                                   const bf16* __restrict__ Vt,
                                                   bf16* __restrict__ Out) {
    __shared__ bf16 lds[4][16 * 32];   // per-wave P staging (S-layout -> A-frag)

    const int lane = threadIdx.x & 31;
    const int wave = threadIdx.x >> 5;
    const int bh   = blockIdx.x;               // 0..B*NH-1
    const int b    = bh >> 4;
    const int h    = bh & 15;
    const int kvh  = h >> 2;                   // N_REP = 4
    const int t0   = (blockIdx.y * 4 + wave) * 16;
    const int hi   = lane >> 4;
    const int l16  = lane & 15;

    const bf16* qbase = Qn + (size_t)(b * NH_ + h)   * T_ * HD_;
    const bf16* kbase = Kn + (size_t)(b * NKV_ + kvh) * T_ * HD_;
    const bf16* vbase = Vt + (size_t)(b * NKV_ + kvh) * HD_ * (size_t)T_;

    // persistent Q fragments: rows t0+l16, 4 chunks of 32 along head dim
    v16bf qf[4];
    {
        const bf16* qrow = qbase + (size_t)(t0 + l16) * HD_;
#pragma unroll
        for (int kc = 0; kc < 4; ++kc) {
            v8bf a0 = *(const v8bf*)(qrow + kc * 32 + hi * 8);
            v8bf a1 = *(const v8bf*)(qrow + kc * 32 + 16 + hi * 8);
            qf[kc] = combine16(a0, a1);
        }
    }

    v8f acc[8];
    float rm[8], rl[8];
#pragma unroll
    for (int j = 0; j < 8; ++j)
#pragma unroll
        for (int v = 0; v < 8; ++v) acc[j][v] = 0.0f;
#pragma unroll
    for (int v = 0; v < 8; ++v) { rm[v] = -INFINITY; rl[v] = 0.0f; }

    const float scale = 0.08838834764831845f;  // 1/sqrt(128)
    bf16* myP = &lds[wave][0];

    for (int s0 = 0; s0 <= t0 + 15; s0 += 32) {
        // ---- scores: two 16x16 subtiles (columns s0..+15 and s0+16..+31)
        v8f sc[2];
#pragma unroll
        for (int sub = 0; sub < 2; ++sub) {
#pragma unroll
            for (int v = 0; v < 8; ++v) sc[sub][v] = 0.0f;
            const bf16* krow = kbase + (size_t)(s0 + sub * 16 + l16) * HD_;
#pragma unroll
            for (int kc = 0; kc < 4; ++kc) {
                v8bf b0 = *(const v8bf*)(krow + kc * 32 + hi * 8);
                v8bf b1 = *(const v8bf*)(krow + kc * 32 + 16 + hi * 8);
                sc[sub] = __builtin_amdgcn_wmma_f32_16x16x32_bf16(
                    false, qf[kc], false, combine16(b0, b1),
                    (short)0, sc[sub], false, false);
            }
        }
        // ---- scale + causal mask (lane holds col, vgpr holds row)
#pragma unroll
        for (int sub = 0; sub < 2; ++sub) {
            int sg = s0 + sub * 16 + l16;
#pragma unroll
            for (int v = 0; v < 8; ++v) {
                int tg = t0 + v + hi * 8;
                float xval = sc[sub][v] * scale;
                sc[sub][v] = (sg > tg) ? -INFINITY : xval;
            }
        }
        // ---- online softmax stats per row (reduce across 16-lane groups)
        float al[8];
#pragma unroll
        for (int v = 0; v < 8; ++v) {
            float m = fmaxf(sc[0][v], sc[1][v]);
#pragma unroll
            for (int off = 8; off >= 1; off >>= 1) m = fmaxf(m, __shfl_xor(m, off, 16));
            float nm = fmaxf(rm[v], m);
            float p0 = __expf(sc[0][v] - nm);
            float p1 = __expf(sc[1][v] - nm);
            float a  = __expf(rm[v] - nm);
            rm[v] = nm;
            sc[0][v] = p0; sc[1][v] = p1;
            float s = p0 + p1;
#pragma unroll
            for (int off = 8; off >= 1; off >>= 1) s += __shfl_xor(s, off, 16);
            rl[v] = rl[v] * a + s;
            al[v] = a;
        }
        // ---- rescale accumulators
#pragma unroll
        for (int j = 0; j < 8; ++j)
#pragma unroll
            for (int v = 0; v < 8; ++v) acc[j][v] *= al[v];

        // ---- stage P (16x32 bf16) through LDS: S-layout -> A-fragment
#pragma unroll
        for (int sub = 0; sub < 2; ++sub)
#pragma unroll
            for (int v = 0; v < 8; ++v)
                myP[(v + hi * 8) * 32 + sub * 16 + l16] = f2bf(sc[sub][v]);
        asm volatile("s_wait_dscnt 0" ::: "memory");
        v8bf p0 = *(const v8bf*)(myP + l16 * 32 + hi * 8);
        v8bf p1 = *(const v8bf*)(myP + l16 * 32 + 16 + hi * 8);
        v16bf pa = combine16(p0, p1);

        // ---- P(16x32) @ V(32x128): 8 n-tiles, B from d-major Vt (contiguous)
#pragma unroll
        for (int j = 0; j < 8; ++j) {
            const bf16* vrow = vbase + (size_t)(j * 16 + l16) * T_ + s0;
            v8bf b0 = *(const v8bf*)(vrow + hi * 8);
            v8bf b1 = *(const v8bf*)(vrow + 16 + hi * 8);
            acc[j] = __builtin_amdgcn_wmma_f32_16x16x32_bf16(
                false, pa, false, combine16(b0, b1),
                (short)0, acc[j], false, false);
        }
    }

    // ---- epilogue: normalize and store bf16 [b][t][h*128 + d]
#pragma unroll
    for (int j = 0; j < 8; ++j) {
#pragma unroll
        for (int v = 0; v < 8; ++v) {
            int t = t0 + v + hi * 8;
            float y = acc[j][v] / rl[v];
            Out[(size_t)(b * T_ + t) * D_ + h * HD_ + j * 16 + l16] = f2bf(y);
        }
    }
}

// ---------------------------------------------------------------------------
// host-side orchestration
// ---------------------------------------------------------------------------
extern "C" void kernel_launch(void* const* d_in, const int* in_sizes, int n_in,
                              void* d_out, int out_size, void* d_ws, size_t ws_size,
                              hipStream_t stream) {
    const float* x       = (const float*)d_in[0];
    const float* wq      = (const float*)d_in[1];
    const float* wk      = (const float*)d_in[2];
    const float* wv      = (const float*)d_in[3];
    const float* wo      = (const float*)d_in[4];
    const float* qnw     = (const float*)d_in[5];
    const float* knw     = (const float*)d_in[6];
    float* out           = (float*)d_out;

    char* ws = (char*)d_ws;
    size_t off = 0;
    auto alloc = [&](size_t bytes) -> char* {
        char* p = ws + off;
        off = (off + bytes + 255) & ~(size_t)255;
        return p;
    };

    bf16* xb   = (bf16*)alloc((size_t)MROWS_ * D_ * 2);
    bf16* wqb  = (bf16*)alloc((size_t)D_ * D_ * 2);
    bf16* wkvb = (bf16*)alloc((size_t)(2 * KVD_) * D_ * 2);   // wk rows 0..511, wv rows 512..1023
    bf16* wob  = (bf16*)alloc((size_t)D_ * D_ * 2);
    float* Qf  = (float*)alloc((size_t)MROWS_ * D_ * 4);
    float* KVf = (float*)alloc((size_t)MROWS_ * (2 * KVD_) * 4); // [m][0..511]=K, [m][512..1023]=V
    bf16* Qn   = (bf16*)alloc((size_t)MROWS_ * D_ * 2);
    bf16* Kn   = (bf16*)alloc((size_t)MROWS_ * KVD_ * 2);
    bf16* Vt   = (bf16*)alloc((size_t)MROWS_ * KVD_ * 2);
    bf16* At   = (bf16*)alloc((size_t)MROWS_ * D_ * 2);

    auto cvt = [&](const float* src, bf16* dst, size_t n) {
        int n4 = (int)(n / 4);
        cvt_f32_bf16<<<(n4 + 255) / 256, 256, 0, stream>>>((const float4*)src, (v4bf*)dst, n4);
    };
    cvt(x,  xb,  (size_t)MROWS_ * D_);
    cvt(wq, wqb, (size_t)D_ * D_);
    cvt(wk, wkvb, (size_t)KVD_ * D_);
    cvt(wv, wkvb + (size_t)KVD_ * D_, (size_t)KVD_ * D_);
    cvt(wo, wob, (size_t)D_ * D_);

    auto gemm = [&](const bf16* A, const bf16* Bw, float* C, int M, int N, int K) {
        gemm_bf16_nt<<<dim3(N / 128, M / 128), 256, 0, stream>>>(A, Bw, C, M, N, K);
    };
    gemm(xb, wqb,  Qf,  MROWS_, D_,        D_);
    gemm(xb, wkvb, KVf, MROWS_, 2 * KVD_,  D_);

    {
        int nvQ = MROWS_ * NH_;    // 65536
        rope_norm_kernel<<<nvQ / 8, 256, 0, stream>>>(Qf, D_, qnw, Qn, NH_, 0, nvQ);
        int nvK = MROWS_ * NKV_;   // 16384
        rope_norm_kernel<<<nvK / 8, 256, 0, stream>>>(KVf, 2 * KVD_, knw, Kn, NKV_, 0, nvK);
        rope_norm_kernel<<<nvK / 8, 256, 0, stream>>>(KVf + KVD_, 2 * KVD_, qnw, Vt, NKV_, 1, nvK);
    }

    attn_kernel<<<dim3(B_ * NH_, T_ / 64), 128, 0, stream>>>(Qn, Kn, Vt, At);

    gemm(At, wob, out, MROWS_, D_, D_);
}
